// SelfAttention2d_63513976373507
// MI455X (gfx1250) — matrix operand order repaired and checked by reading, not compiled, not run on hardware
//
#include <hip/hip_runtime.h>

typedef __attribute__((ext_vector_type(16))) _Float16 v16h;
typedef __attribute__((ext_vector_type(8)))  float    v8f;
typedef __attribute__((ext_vector_type(4)))  unsigned int u32x4;
typedef __attribute__((ext_vector_type(4)))  float    f32x4;

namespace {
constexpr int Bsz = 4;
constexpr int C   = 256;
constexpr int Tn  = 4096;    // H*W tokens
constexpr int NH  = 4;
constexpr int HD  = 64;      // head dim
constexpr int GRP = 32;
constexpr int CPG = C / GRP; // 8
constexpr int C3  = 3 * C;   // 768

union H16 { v16h v; _Float16 h[16]; unsigned u[8]; u32x4 q[2]; };
union F8  { v8f  v; float    f[8];  f32x4 g[2]; };
union HV8 { u32x4 q; _Float16 h[8]; };
union HV4 { unsigned long long u; _Float16 h[4]; };
} // namespace

// ---- CDNA5 async global->LDS copy (16B per lane), tracked by ASYNCcnt ----
__device__ __forceinline__ void async_copy_b128(void* lds_dst, const void* gsrc) {
  unsigned lds = (unsigned)(unsigned long long)lds_dst;  // low 32b = LDS offset
  unsigned long long ga = (unsigned long long)gsrc;
  asm volatile("global_load_async_to_lds_b128 %0, %1, off"
               :: "v"(lds), "v"(ga) : "memory");
}
__device__ __forceinline__ void wait_async() {
  asm volatile("s_wait_asynccnt 0x0" ::: "memory");
}

__device__ __forceinline__ v8f wmma_f16(const H16& a, const H16& b, v8f c) {
  return __builtin_amdgcn_wmma_f32_16x16x32_f16(
      false, a.v, false, b.v, (short)0, c, false, false);
}

// A 16x32 f16 (row-major [M][K] tile): two 16B LDS loads per lane.
__device__ __forceinline__ void load_a_frag(H16& a, const _Float16* row, int half) {
  a.q[0] = *(const u32x4*)(row + 8 * half);
  a.q[1] = *(const u32x4*)(row + 16 + 8 * half);
}
// B 32x16 f16 from column-major [N][K] tile: two 16B LDS loads per lane.
__device__ __forceinline__ void load_b_frag_t(H16& b, const _Float16* colrow, int half) {
  b.q[0] = *(const u32x4*)(colrow + 8 * half);
  b.q[1] = *(const u32x4*)(colrow + 16 + 8 * half);
}

// ---------------- Kernel 1: GroupNorm statistics ----------------
__global__ __launch_bounds__(256) void gn_stats_kernel(const float* __restrict__ x,
                                                       float* __restrict__ stats) {
  int bg = blockIdx.x;
  const float* base = x + (size_t)bg * CPG * Tn;
  float s = 0.f, ss = 0.f;
  for (int i = threadIdx.x; i < CPG * Tn; i += 256) {
    float v = base[i];
    s += v; ss += v * v;
  }
#pragma unroll
  for (int m = 1; m < 32; m <<= 1) {
    s  += __shfl_xor(s,  m, 32);
    ss += __shfl_xor(ss, m, 32);
  }
  __shared__ float ps[8], pss[8];
  int wave = threadIdx.x >> 5, lane = threadIdx.x & 31;
  if (lane == 0) { ps[wave] = s; pss[wave] = ss; }
  __syncthreads();
  if (threadIdx.x == 0) {
    float S = 0.f, SS = 0.f;
    for (int w = 0; w < 8; ++w) { S += ps[w]; SS += pss[w]; }
    const float inv = 1.0f / (float)(CPG * Tn);
    float mean = S * inv;
    float var  = SS * inv - mean * mean;
    stats[2 * bg]     = mean;
    stats[2 * bg + 1] = rsqrtf(var + 1e-5f);
  }
}

// -------- Kernel 2: GroupNorm apply -> f16 hidden, token-major [b][t][c] --------
__global__ __launch_bounds__(256) void gn_apply_kernel(const float* __restrict__ x,
                                                       const float* __restrict__ stats,
                                                       const float* __restrict__ gw,
                                                       const float* __restrict__ gb,
                                                       _Float16* __restrict__ hidden) {
  size_t idx = (size_t)blockIdx.x * 256 + threadIdx.x;  // b*2^20 + c*2^12 + t
  int t = (int)(idx & (Tn - 1));
  int c = (int)((idx >> 12) & (C - 1));
  int b = (int)(idx >> 20);
  int bg = b * GRP + (c >> 3);
  float mean = stats[2 * bg], rstd = stats[2 * bg + 1];
  float v = (x[idx] - mean) * rstd * gw[c] + gb[c];
  hidden[((size_t)b * Tn + t) * C + c] = (_Float16)v;
}

// -------- Kernel 3: QKV GEMM (double-buffered): qkv[b][t][o], v -> vT[b][c'][t] --------
__global__ __launch_bounds__(256) void qkv_gemm_kernel(const float* __restrict__ w,
                                                       const float* __restrict__ bias,
                                                       const _Float16* __restrict__ hidden,
                                                       _Float16* __restrict__ qkv,
                                                       _Float16* __restrict__ vT) {
  int b  = blockIdx.z;
  int tb = blockIdx.y * 64;
  int ob = blockIdx.x * 64;
  __shared__ _Float16 As[2][64][32];   // [t][c]
  __shared__ _Float16 Bw[2][64][32];   // [o][c] (column-major for WMMA B)
  int tid = threadIdx.x;
  int wave = tid >> 5, lane = tid & 31, lane16 = lane & 15, half = lane >> 4;
  int mi = wave >> 1, ni = wave & 1;
  const _Float16* hb = hidden + (size_t)b * Tn * C;
  int ra = tid >> 2, ca = (tid & 3) * 8;            // one async op per thread
  F8 c0 = {}, c1 = {};

  async_copy_b128(&As[0][ra][ca], hb + (size_t)(tb + ra) * C + ca);
  for (int j = tid; j < 512; j += 256) {
    int r = j >> 3, cb = j & 7;
    f32x4 wv = *(const f32x4*)(w + (size_t)(ob + r) * C + cb * 4);
    HV4 h4;
#pragma unroll
    for (int i = 0; i < 4; ++i) h4.h[i] = (_Float16)wv[i];
    *(unsigned long long*)&Bw[0][r][cb * 4] = h4.u;
  }
  wait_async();
  __syncthreads();

  for (int i = 0; i < 8; ++i) {
    int cur = i & 1;
    if (i < 7) {                                    // prefetch next K-tile
      int kk = (i + 1) * 32, nxt = cur ^ 1;
      async_copy_b128(&As[nxt][ra][ca], hb + (size_t)(tb + ra) * C + kk + ca);
      for (int j = tid; j < 512; j += 256) {
        int r = j >> 3, cb = j & 7;
        f32x4 wv = *(const f32x4*)(w + (size_t)(ob + r) * C + kk + cb * 4);
        HV4 h4;
#pragma unroll
        for (int k = 0; k < 4; ++k) h4.h[k] = (_Float16)wv[k];
        *(unsigned long long*)&Bw[nxt][r][cb * 4] = h4.u;
      }
    }
    H16 a, b0, b1;
    load_a_frag(a, &As[cur][mi * 16 + lane16][0], half);
    load_b_frag_t(b0, &Bw[cur][ni * 32 + lane16][0], half);
    load_b_frag_t(b1, &Bw[cur][ni * 32 + 16 + lane16][0], half);
    c0.v = wmma_f16(a, b0, c0.v);
    c1.v = wmma_f16(a, b1, c1.v);
    if (i < 7) { wait_async(); __syncthreads(); }
  }

  int o0c = ob + ni * 32 + lane16, o1c = o0c + 16;
  float bv0 = bias[o0c], bv1 = bias[o1c];
  int tb8 = tb + mi * 16 + 8 * half;                // lane's 8 consecutive t rows
  if (ob < C) {                                     // q: fold in 1/sqrt(hd)
    _Float16* qb = qkv + (size_t)b * Tn * C3;
#pragma unroll
    for (int r = 0; r < 8; ++r) {
      size_t row = (size_t)(tb8 + r) * C3;
      qb[row + o0c] = (_Float16)((c0.f[r] + bv0) * 0.125f);
      qb[row + o1c] = (_Float16)((c1.f[r] + bv1) * 0.125f);
    }
  } else if (ob < 2 * C) {                          // k
    _Float16* qb = qkv + (size_t)b * Tn * C3;
#pragma unroll
    for (int r = 0; r < 8; ++r) {
      size_t row = (size_t)(tb8 + r) * C3;
      qb[row + o0c] = (_Float16)(c0.f[r] + bv0);
      qb[row + o1c] = (_Float16)(c1.f[r] + bv1);
    }
  } else {                                          // v: channel-major, packed 16B store
    _Float16* vtb = vT + (size_t)b * C * Tn;
    HV8 p0, p1;
#pragma unroll
    for (int r = 0; r < 8; ++r) {
      p0.h[r] = (_Float16)(c0.f[r] + bv0);
      p1.h[r] = (_Float16)(c1.f[r] + bv1);
    }
    *(u32x4*)(vtb + (size_t)(o0c - 2 * C) * Tn + tb8) = p0.q;
    *(u32x4*)(vtb + (size_t)(o1c - 2 * C) * Tn + tb8) = p1.q;
  }
}

// ---------------- Kernel 4: fused flash attention (double-buffered K/V) ----------------
// grid (Tn/128, B*NH); 8 waves, wave w owns query rows [w*16, w*16+16).
__global__ __launch_bounds__(256) void attn_kernel(const _Float16* __restrict__ qkv,
                                                   const _Float16* __restrict__ vT,
                                                   _Float16* __restrict__ att) {
  constexpr int BM = 128, BN = 64, NB = Tn / BN;
  int bh = blockIdx.y;
  int b = bh >> 2, h = bh & 3;
  int t0 = blockIdx.x * BM;
  const _Float16* qg = qkv + (size_t)b * Tn * C3 + h * HD;   // q[t][d] (pre-scaled)
  const _Float16* kg = qg + C;                               // k[t][d]
  const _Float16* vg = vT + ((size_t)b * C + h * HD) * Tn;   // v[d][t]

  // SQP: Q tile [t][d] during prologue; per-wave P staging afterwards
  // (wave w only ever touches rows [w*16, w*16+16) in both roles -> safe overlay)
  __shared__ _Float16 SQP[BM][HD];       // 16 KB
  __shared__ _Float16 KsT[2][BN][HD];    // 16 KB  [s][d] col-major B (n=s,k=d)
  __shared__ _Float16 Vs [2][HD][BN];    // 16 KB  [d][s] col-major B (n=d,k=s)

  int tid = threadIdx.x;
  int wave = tid >> 5, lane = tid & 31, lane16 = lane & 15, half = lane >> 4;

  for (int j = tid; j < BM * 8; j += 256) {            // Q
    int tl = j >> 3, db = j & 7;
    async_copy_b128(&SQP[tl][db * 8], qg + (size_t)(t0 + tl) * C3 + db * 8);
  }
  for (int j = tid; j < BN * 8; j += 256) {            // K block 0
    int s = j >> 3, db = j & 7;
    async_copy_b128(&KsT[0][s][db * 8], kg + (size_t)s * C3 + db * 8);
  }
  for (int j = tid; j < BN * 8; j += 256) {            // V block 0 (straight copy!)
    int d = j >> 3, sb = j & 7;
    async_copy_b128(&Vs[0][d][sb * 8], vg + (size_t)d * Tn + sb * 8);
  }
  wait_async();
  __syncthreads();

  H16 qa[2];
#pragma unroll
  for (int ks = 0; ks < 2; ++ks)
    load_a_frag(qa[ks], &SQP[wave * 16 + lane16][ks * 32], half);

  F8 o0 = {}, o1 = {}, o2 = {}, o3 = {};
  float mrow[8], lrow[8];
#pragma unroll
  for (int r = 0; r < 8; ++r) { mrow[r] = -1e30f; lrow[r] = 0.f; }

  for (int ib = 0; ib < NB; ++ib) {
    int cur = ib & 1;
    if (ib + 1 < NB) {                                 // prefetch next K/V block
      int s1 = (ib + 1) * BN, nxt = cur ^ 1;
      for (int j = tid; j < BN * 8; j += 256) {
        int s = j >> 3, db = j & 7;
        async_copy_b128(&KsT[nxt][s][db * 8], kg + (size_t)(s1 + s) * C3 + db * 8);
      }
      for (int j = tid; j < BN * 8; j += 256) {
        int d = j >> 3, sb = j & 7;
        async_copy_b128(&Vs[nxt][d][sb * 8], vg + (size_t)d * Tn + s1 + sb * 8);
      }
    }

    // S = (q*scale) K^T : 16(t) x 64(s)
    F8 sc[4] = {};
#pragma unroll
    for (int jn = 0; jn < 4; ++jn) {
#pragma unroll
      for (int ks = 0; ks < 2; ++ks) {
        H16 bf;
        load_b_frag_t(bf, &KsT[cur][jn * 16 + lane16][ks * 32], half);
        sc[jn].v = wmma_f16(qa[ks], bf, sc[jn].v);
      }
    }

    // online softmax; C-layout row (r + 8*half) lives in one 16-lane half
#pragma unroll
    for (int r = 0; r < 8; ++r) {
      float mx = fmaxf(fmaxf(sc[0].f[r], sc[1].f[r]), fmaxf(sc[2].f[r], sc[3].f[r]));
#pragma unroll
      for (int msk = 1; msk < 16; msk <<= 1)
        mx = fmaxf(mx, __shfl_xor(mx, msk, 32));
      float mnew  = fmaxf(mrow[r], mx);
      float alpha = __expf(mrow[r] - mnew);
      mrow[r] = mnew;
      float psum = 0.f;
#pragma unroll
      for (int jn = 0; jn < 4; ++jn) {
        float p = __expf(sc[jn].f[r] - mnew);
        psum += p;
        SQP[wave * 16 + r + 8 * half][jn * 16 + lane16] = (_Float16)p;
      }
#pragma unroll
      for (int msk = 1; msk < 16; msk <<= 1)
        psum += __shfl_xor(psum, msk, 32);
      lrow[r] = lrow[r] * alpha + psum;
      o0.f[r] *= alpha; o1.f[r] *= alpha; o2.f[r] *= alpha; o3.f[r] *= alpha;
    }

    // O += P V (contraction over s; same-wave LDS ops are in-order)
#pragma unroll
    for (int ks = 0; ks < 2; ++ks) {
      H16 pa, vf;
      load_a_frag(pa, &SQP[wave * 16 + lane16][ks * 32], half);
      load_b_frag_t(vf, &Vs[cur][0 * 16 + lane16][ks * 32], half);
      o0.v = wmma_f16(pa, vf, o0.v);
      load_b_frag_t(vf, &Vs[cur][1 * 16 + lane16][ks * 32], half);
      o1.v = wmma_f16(pa, vf, o1.v);
      load_b_frag_t(vf, &Vs[cur][2 * 16 + lane16][ks * 32], half);
      o2.v = wmma_f16(pa, vf, o2.v);
      load_b_frag_t(vf, &Vs[cur][3 * 16 + lane16][ks * 32], half);
      o3.v = wmma_f16(pa, vf, o3.v);
    }

    if (ib + 1 < NB) { wait_async(); __syncthreads(); }
  }

  // normalize, store attended token-major: att[b][t][h*64+d]
  int tb8 = t0 + wave * 16 + 8 * half;
#pragma unroll
  for (int r = 0; r < 8; ++r) {
    float inv = 1.0f / lrow[r];
    size_t row = ((size_t)b * Tn + tb8 + r) * C + h * HD;
    att[row + 0 * 16 + lane16] = (_Float16)(o0.f[r] * inv);
    att[row + 1 * 16 + lane16] = (_Float16)(o1.f[r] * inv);
    att[row + 2 * 16 + lane16] = (_Float16)(o2.f[r] * inv);
    att[row + 3 * 16 + lane16] = (_Float16)(o3.f[r] * inv);
  }
}

// ------ Kernel 5: proj GEMM (double-buffered) + bias + residual, fp32 out[b][o][t] ------
__global__ __launch_bounds__(256) void proj_gemm_kernel(const float* __restrict__ w,
                                                        const float* __restrict__ bias,
                                                        const _Float16* __restrict__ att,
                                                        const float* __restrict__ xin,
                                                        float* __restrict__ out) {
  int b  = blockIdx.z;
  int tb = blockIdx.y * 64;
  int ob = blockIdx.x * 64;
  __shared__ _Float16 As[2][64][32];
  __shared__ _Float16 Bw[2][64][32];
  int tid = threadIdx.x;
  int wave = tid >> 5, lane = tid & 31, lane16 = lane & 15, half = lane >> 4;
  int mi = wave >> 1, ni = wave & 1;
  const _Float16* ab = att + (size_t)b * Tn * C;
  int ra = tid >> 2, ca = (tid & 3) * 8;
  F8 c0 = {}, c1 = {};

  async_copy_b128(&As[0][ra][ca], ab + (size_t)(tb + ra) * C + ca);
  for (int j = tid; j < 512; j += 256) {
    int r = j >> 3, cb = j & 7;
    f32x4 wv = *(const f32x4*)(w + (size_t)(ob + r) * C + cb * 4);
    HV4 h4;
#pragma unroll
    for (int i = 0; i < 4; ++i) h4.h[i] = (_Float16)wv[i];
    *(unsigned long long*)&Bw[0][r][cb * 4] = h4.u;
  }
  wait_async();
  __syncthreads();

  for (int i = 0; i < 8; ++i) {
    int cur = i & 1;
    if (i < 7) {
      int kk = (i + 1) * 32, nxt = cur ^ 1;
      async_copy_b128(&As[nxt][ra][ca], ab + (size_t)(tb + ra) * C + kk + ca);
      for (int j = tid; j < 512; j += 256) {
        int r = j >> 3, cb = j & 7;
        f32x4 wv = *(const f32x4*)(w + (size_t)(ob + r) * C + kk + cb * 4);
        HV4 h4;
#pragma unroll
        for (int k = 0; k < 4; ++k) h4.h[k] = (_Float16)wv[k];
        *(unsigned long long*)&Bw[nxt][r][cb * 4] = h4.u;
      }
    }
    H16 a, b0, b1;
    load_a_frag(a, &As[cur][mi * 16 + lane16][0], half);
    load_b_frag_t(b0, &Bw[cur][ni * 32 + lane16][0], half);
    load_b_frag_t(b1, &Bw[cur][ni * 32 + 16 + lane16][0], half);
    c0.v = wmma_f16(a, b0, c0.v);
    c1.v = wmma_f16(a, b1, c1.v);
    if (i < 7) { wait_async(); __syncthreads(); }
  }

  // epilogue: each lane owns 8 consecutive t rows at two o columns -> f32x4 ops
  int o0c = ob + ni * 32 + lane16, o1c = o0c + 16;
  float bv0 = bias[o0c], bv1 = bias[o1c];
  int tb8 = tb + mi * 16 + 8 * half;
  size_t off0 = ((size_t)b * C + o0c) * Tn + tb8;
  size_t off1 = ((size_t)b * C + o1c) * Tn + tb8;
  *(f32x4*)(out + off0)     = *(const f32x4*)(xin + off0)     + c0.g[0] + bv0;
  *(f32x4*)(out + off0 + 4) = *(const f32x4*)(xin + off0 + 4) + c0.g[1] + bv0;
  *(f32x4*)(out + off1)     = *(const f32x4*)(xin + off1)     + c1.g[0] + bv1;
  *(f32x4*)(out + off1 + 4) = *(const f32x4*)(xin + off1 + 4) + c1.g[1] + bv1;
}

extern "C" void kernel_launch(void* const* d_in, const int* in_sizes, int n_in,
                              void* d_out, int out_size, void* d_ws, size_t ws_size,
                              hipStream_t stream) {
  (void)in_sizes; (void)n_in; (void)out_size; (void)ws_size;
  const float* x      = (const float*)d_in[0];
  const float* gn_w   = (const float*)d_in[1];
  const float* gn_b   = (const float*)d_in[2];
  const float* qkv_w  = (const float*)d_in[3];
  const float* qkv_b  = (const float*)d_in[4];
  const float* proj_w = (const float*)d_in[5];
  const float* proj_b = (const float*)d_in[6];
  float* out = (float*)d_out;

  // ws: hidden f16 [b][t][c] (8MB) | qkv f16 [b][t][3c] (24MB) |
  //     att f16 [b][t][c] (8MB) | vT f16 [b][c][t] (8MB) | gn stats (1KB)
  _Float16* hidden = (_Float16*)d_ws;
  _Float16* qkv    = hidden + (size_t)Bsz * Tn * C;
  _Float16* att    = qkv + (size_t)Bsz * Tn * C3;
  _Float16* vT     = att + (size_t)Bsz * Tn * C;
  float*    stats  = (float*)(vT + (size_t)Bsz * C * Tn);

  gn_stats_kernel<<<Bsz * GRP, 256, 0, stream>>>(x, stats);
  gn_apply_kernel<<<(Bsz * C * Tn) / 256, 256, 0, stream>>>(x, stats, gn_w, gn_b, hidden);
  qkv_gemm_kernel<<<dim3(C3 / 64, Tn / 64, Bsz), 256, 0, stream>>>(qkv_w, qkv_b, hidden, qkv, vT);
  attn_kernel<<<dim3(Tn / 128, Bsz * NH), 256, 0, stream>>>(qkv, vT, att);
  proj_gemm_kernel<<<dim3(C / 64, Tn / 64, Bsz), 256, 0, stream>>>(proj_w, proj_b, att, x, out);
}
